// TPA_13383118094840
// MI455X (gfx1250) — matrix operand order repaired and checked by reading, not compiled, not run on hardware
//
#include <hip/hip_runtime.h>

// ---------------------------------------------------------------------------
// Types for CDNA5 WMMA (wave32)
// ---------------------------------------------------------------------------
typedef __attribute__((ext_vector_type(16))) __bf16 v16bf;
typedef __attribute__((ext_vector_type(8)))  float  v8f;
typedef __attribute__((ext_vector_type(4)))  unsigned int u32x4;
typedef __attribute__((ext_vector_type(8)))  int          i32x8;
typedef __attribute__((ext_vector_type(4)))  int          i32x4;

// Problem constants (match reference)
constexpr int  Bn  = 4;
constexpr int  Sn  = 2048;
constexpr int  Dn  = 2048;
constexpr int  Hn  = 16;
constexpr int  DHn = 128;
constexpr int  RQn = 6, RKn = 2, RVn = 2;
constexpr int  NP  = RQn*Hn + RKn*Hn + RVn*Hn + RQn*DHn + RKn*DHn + RVn*DHn; // 1440
constexpr int  NPP = 1536;            // NP padded to a multiple of 128
constexpr long MR  = (long)Bn * Sn;   // 8192 token rows

// Column offsets inside the fused projection output row (length NP)
constexpr int OFF_AQ = 0;                // RQ*H  = 96
constexpr int OFF_AK = OFF_AQ + RQn*Hn;  // 96
constexpr int OFF_AV = OFF_AK + RKn*Hn;  // 128
constexpr int OFF_BQ = OFF_AV + RVn*Hn;  // 160
constexpr int OFF_BK = OFF_BQ + RQn*DHn; // 928
constexpr int OFF_BV = OFF_BK + RKn*DHn; // 1184

// Tensor Data Mover availability (device pass only; arity differs by clang)
#if defined(__HIP_DEVICE_COMPILE__) && __has_builtin(__builtin_amdgcn_tensor_load_to_lds)
#define USE_TDM 1
#else
#define USE_TDM 0
#endif

// ---------------------------------------------------------------------------
// WMMA helpers (layouts per CDNA5 ISA §7.12.2)
// ---------------------------------------------------------------------------
__device__ __forceinline__ v8f wmma_bf16(v16bf a, v16bf b, v8f c) {
  return __builtin_amdgcn_wmma_f32_16x16x32_bf16(false, a, false, b, (short)0, c,
                                                 false, false);
}

// A-fragment: 16x32 bf16 tile, row-major with stride ld (elements).
// lane holds row (lane&15); K = kb..kb+7 and kb+16..kb+23, kb = (lane>>4)*8.
// Requires 16B-aligned rows (ld*2 % 16 == 0).
__device__ __forceinline__ v16bf load_a16(const __bf16* base, int ld) {
  int lane = threadIdx.x & 31;
  int r  = lane & 15;
  int kb = (lane >> 4) << 3;
  const __bf16* p = base + r * ld + kb;
  union { v16bf v; uint4 q[2]; } u;
  u.q[0] = *(const uint4*)p;         // elements 0..7
  u.q[1] = *(const uint4*)(p + 16);  // elements 8..15
  return u.v;
}

// B-fragment from a K-contiguous layout: element (k, n) at base + n*ld + k.
// lane group g = (lane>>4) holds K = g*8..g*8+7 and 16+g*8..16+g*8+7; column = lane&15.
__device__ __forceinline__ v16bf load_b16k(const __bf16* base, int ld) {
  int lane = threadIdx.x & 31;
  int c  = lane & 15;
  int kb = (lane >> 4) << 3;
  const __bf16* p = base + c * ld + kb;
  union { v16bf v; uint4 q[2]; } u;
  u.q[0] = *(const uint4*)p;
  u.q[1] = *(const uint4*)(p + 16);
  return u.v;
}

// ---------------------------------------------------------------------------
// TDM: async 2D bf16 tile load into LDS (descriptor per CDNA5 ISA).
// tile_d0 elements per row, tile_d1 rows, row stride stride_d0 elements.
// LDS pad feature: insert (pad_acode+1) dwords every 2^(pad_icode+1) dwords.
// ---------------------------------------------------------------------------
__device__ __forceinline__ void tdm_load_2d(unsigned lds_off, const void* gsrc,
                                            unsigned tile_d0, unsigned tile_d1,
                                            unsigned stride_d0,
                                            unsigned pad_icode,
                                            unsigned pad_acode) {
#if USE_TDM
  unsigned long long ga = (unsigned long long)gsrc;
  u32x4 g0;
  g0[0] = 1u;                                   // count = 1 (valid descriptor)
  g0[1] = lds_off;                              // lds_addr (bytes)
  g0[2] = (unsigned)ga;                         // global_addr[31:0]
  g0[3] = (unsigned)((ga >> 32) & 0x01ffffffu)  // global_addr[56:32]
          | 0x80000000u;                        // type = 2 ("image")
  i32x8 g1;
  g1[0] = (int)((1u << 16)                      // data_size = 2 bytes
              | (1u << 20)                      // pad_enable
              | (pad_icode << 22)               // pad_interval
              | (pad_acode << 25));             // pad_amount
  g1[1] = (int)(tile_d0 << 16);                 // tensor_dim0 = tile_d0
  g1[2] = (int)(tile_d1 << 16);                 // tensor_dim1 = tile_d1
  g1[3] = (int)(tile_d0 << 16);                 // tile_dim0
  g1[4] = (int)(tile_d1 & 0xffffu);             // tile_dim1 ; tile_dim2 = 0
  g1[5] = (int)stride_d0;                       // tensor_dim0_stride[31:0]
  g1[6] = 0;
  g1[7] = 0;
  i32x4 gz = {0, 0, 0, 0};
#if defined(__clang_major__) && __clang_major__ >= 23
  i32x8 gz8 = {0, 0, 0, 0, 0, 0, 0, 0};
  __builtin_amdgcn_tensor_load_to_lds(g0, g1, gz, gz, gz8, 0);
#else
  __builtin_amdgcn_tensor_load_to_lds(g0, g1, gz, gz, 0);
#endif
#else
  (void)lds_off; (void)gsrc; (void)tile_d0; (void)tile_d1;
  (void)stride_d0; (void)pad_icode; (void)pad_acode;
#endif
}

// ---------------------------------------------------------------------------
// Elementwise conversion / packing / fill
// ---------------------------------------------------------------------------
__global__ void cvt_f32_bf16(const float* __restrict__ s, __bf16* __restrict__ d,
                             long n) {
  long i = (long)blockIdx.x * blockDim.x + threadIdx.x;
  if (i < n) d[i] = (__bf16)s[i];
}

__global__ void fill_zero_bf16(__bf16* __restrict__ p, long n) {
  long i = (long)blockIdx.x * blockDim.x + threadIdx.x;
  if (i < n) p[i] = (__bf16)0.f;
}

__global__ void pack_w(const float* __restrict__ w, __bf16* __restrict__ wcat,
                       int cols, int off) {
  long n = (long)Dn * cols;
  long i = (long)blockIdx.x * blockDim.x + threadIdx.x;
  if (i < n) {
    long r = i / cols;
    int  c = (int)(i - r * cols);
    wcat[r * NPP + off + c] = (__bf16)w[i];
  }
}

// ---------------------------------------------------------------------------
// bf16 GEMM, f32 output.  C[M,N] = A[M,K] * B[K,N]
// Block tile 128x128, BK=32, 256 threads = 8 waves (4m x 2n wave tiles of
// 32x64 -> 8 WMMAs per wave per k-step). A tile arrives via the Tensor Data
// Mover (async DMA, padded LDS); B tile staged transposed so B-fragments are
// K-contiguous. M, N multiples of 128; K multiple of 32. No guards.
// ---------------------------------------------------------------------------
__global__ void __launch_bounds__(256)
gemm_bf16_f32(const __bf16* __restrict__ A, const __bf16* __restrict__ Bm,
              float* __restrict__ C, int M, int N, int K) {
  __shared__ __bf16 As[128][40];   // [m][k]  row = 80B (16B aligned)
  __shared__ __bf16 Bst[128][40];  // [n][k]  transposed

  const int tid  = threadIdx.x;
  const int lane = tid & 31;
  const int wave = tid >> 5;
  const int wm   = (wave >> 1) * 32;  // 0,32,64,96
  const int wn   = (wave & 1) * 64;   // 0,64
  const int m0   = blockIdx.y * 128;
  const int n0   = blockIdx.x * 128;
  const unsigned asOff = (unsigned)(unsigned long long)&As[0][0];

  const v8f vzero = {0.f, 0.f, 0.f, 0.f, 0.f, 0.f, 0.f, 0.f};
  v8f c[2][4];
#pragma unroll
  for (int i = 0; i < 2; ++i)
#pragma unroll
    for (int j = 0; j < 4; ++j) c[i][j] = vzero;

  for (int k0 = 0; k0 < K; k0 += 32) {
    __syncthreads();  // previous fragments fully consumed
#if USE_TDM
    if (wave == 0)
      tdm_load_2d(asOff, A + (long)m0 * K + k0, /*d0=*/32, /*d1=*/128,
                  /*stride=*/(unsigned)K, /*pad 16dw*/3, /*+4dw*/3);
#else
    for (int i = tid; i < 512; i += 256) {
      int ar = i >> 2;
      int ac = (i & 3) << 3;
      *(uint4*)&As[ar][ac] = *(const uint4*)&A[(long)(m0 + ar) * K + k0 + ac];
    }
#endif
    // B tile: 32x128 bf16, vector load + transposed scalar stores
    for (int i = tid; i < 512; i += 256) {
      int kr = i >> 4;
      int nc = (i & 15) << 3;
      union { uint4 q; __bf16 e[8]; } t;
      t.q = *(const uint4*)&Bm[(long)(k0 + kr) * N + n0 + nc];
#pragma unroll
      for (int j = 0; j < 8; ++j) Bst[nc + j][kr] = t.e[j];
    }
#if USE_TDM
    if (wave == 0) __builtin_amdgcn_s_wait_tensorcnt(0);
#endif
    __syncthreads();

    v16bf a0 = load_a16(&As[wm][0], 40);
    v16bf a1 = load_a16(&As[wm + 16][0], 40);
#pragma unroll
    for (int j = 0; j < 4; ++j) {
      v16bf bj = load_b16k(&Bst[wn + 16 * j][0], 40);
      c[0][j] = wmma_bf16(a0, bj, c[0][j]);
      c[1][j] = wmma_bf16(a1, bj, c[1][j]);
    }
  }

  const int rb = (lane >> 4) * 8;
  const int cc = lane & 15;
#pragma unroll
  for (int ci = 0; ci < 2; ++ci)
#pragma unroll
    for (int j = 0; j < 4; ++j)
#pragma unroll
      for (int i = 0; i < 8; ++i) {
        long row = m0 + wm + ci * 16 + rb + i;
        C[row * N + n0 + wn + j * 16 + cc] = c[ci][j][i];
      }
}

// ---------------------------------------------------------------------------
// RoPE (fixed position index S=2048, exactly as the reference) + rank
// contraction. One block per token. Q/K/V out in (B,H,S,DH) bf16 with the
// 1/R scaling folded in. Projection rows have stride NPP.
// ---------------------------------------------------------------------------
__global__ void __launch_bounds__(256)
rope_qkv(const float* __restrict__ P, const float* __restrict__ cf,
         __bf16* __restrict__ Qo, __bf16* __restrict__ Ko,
         __bf16* __restrict__ Vo) {
  __shared__ float sP[NP];
  __shared__ float sfr[DHn / 2];
  __shared__ float sfi[DHn / 2];

  const int tid = threadIdx.x;
  const long bs = blockIdx.x;          // b*S + s
  const int b   = (int)(bs >> 11);     // S = 2048
  const int s   = (int)(bs & 2047);

  const float* row = P + bs * NPP;
  for (int i = tid; i < NP; i += 256) sP[i] = row[i];
  if (tid < DHn / 2) {
    // complex_freq flat index: pos*DH + 2*j (+1);  pos = S = 2048
    sfr[tid] = cf[(long)Sn * DHn + 2 * tid];
    sfi[tid] = cf[(long)Sn * DHn + 2 * tid + 1];
  }
  __syncthreads();

  for (int idx = tid; idx < Hn * DHn; idx += 256) {
    const int h = idx >> 7;
    const int d = idx & 127;
    const int j = d >> 1;
    const bool odd = d & 1;
    const float fr = sfr[j], fi = sfi[j];

    float q = 0.f, k = 0.f, v = 0.f;
#pragma unroll
    for (int r = 0; r < RQn; ++r) {
      float x0 = sP[OFF_BQ + r * DHn + 2 * j];
      float x1 = sP[OFF_BQ + r * DHn + 2 * j + 1];
      float rb = odd ? (x0 * fi + x1 * fr) : (x0 * fr - x1 * fi);
      q += sP[OFF_AQ + r * Hn + h] * rb;
    }
#pragma unroll
    for (int r = 0; r < RKn; ++r) {
      float x0 = sP[OFF_BK + r * DHn + 2 * j];
      float x1 = sP[OFF_BK + r * DHn + 2 * j + 1];
      float rb = odd ? (x0 * fi + x1 * fr) : (x0 * fr - x1 * fi);
      k += sP[OFF_AK + r * Hn + h] * rb;
      v += sP[OFF_AV + r * Hn + h] * sP[OFF_BV + r * DHn + d];
    }
    const long o = (((long)(b * Hn + h)) * Sn + s) * DHn + d;
    Qo[o] = (__bf16)(q * (1.f / RQn));
    Ko[o] = (__bf16)(k * (1.f / RKn));
    Vo[o] = (__bf16)(v * (1.f / RVn));
  }
}

// ---------------------------------------------------------------------------
// Flash attention. Grid: b * H * (S/128) blocks of 256 threads (8 waves).
// Each wave owns 16 q rows. K tile (32x128) arrives via the Tensor Data Mover
// (async DMA into padded LDS); V tile is staged transposed so PV B-fragments
// are K-contiguous. O written as (B*S, D) bf16, column = h*DH + d.
// ---------------------------------------------------------------------------
__global__ void __launch_bounds__(256)
flash_attn(const __bf16* __restrict__ Q, const __bf16* __restrict__ K,
           const __bf16* __restrict__ V, __bf16* __restrict__ O) {
  __shared__ __bf16 Kt[32][136];    // [key][d]   272B rows
  __shared__ __bf16 Vt[128][40];    // [d][key]   transposed, 80B rows
  __shared__ __bf16 Pst[8][16][40]; // per-wave P staging (C->A relayout)

  const int bid  = blockIdx.x;
  const int qblk = bid & 15;          // S/128 = 16
  const int h    = (bid >> 4) & 15;
  const int b    = bid >> 8;

  const int tid  = threadIdx.x;
  const int lane = tid & 31;
  const int wave = tid >> 5;
  const int rb   = (lane >> 4) * 8;   // C-layout row base
  const int cc   = lane & 15;         // C-layout column

  const long headBase = (long)(b * Hn + h) * Sn;
  const long qrow0    = headBase + qblk * 128 + wave * 16;
  const unsigned ktOff = (unsigned)(unsigned long long)&Kt[0][0];

  // Q fragments: 16 rows x 128 as 4 A-frags of 16x32
  v16bf qf[4];
#pragma unroll
  for (int kc = 0; kc < 4; ++kc)
    qf[kc] = load_a16(Q + qrow0 * DHn + kc * 32, DHn);

  const v8f vzero = {0.f, 0.f, 0.f, 0.f, 0.f, 0.f, 0.f, 0.f};
  v8f acc[8];
#pragma unroll
  for (int i = 0; i < 8; ++i) acc[i] = vzero;
  float m8[8], l8[8];
#pragma unroll
  for (int i = 0; i < 8; ++i) { m8[i] = -3.0e38f; l8[i] = 0.f; }

  const float scale = 0.08838834764831845f;  // 1/sqrt(128)

  for (int t0 = 0; t0 < Sn; t0 += 32) {
    __syncthreads();  // previous tiles fully consumed
#if USE_TDM
    if (wave == 0)
      tdm_load_2d(ktOff, K + (headBase + t0) * DHn, /*d0=*/128, /*d1=*/32,
                  /*stride=*/128, /*pad 64dw*/5, /*+4dw*/3);
#else
    for (int i = tid; i < 512; i += 256) {
      int r  = i >> 4;
      int c8 = (i & 15) << 3;
      *(uint4*)&Kt[r][c8] = *(const uint4*)&K[(headBase + t0 + r) * DHn + c8];
    }
#endif
    // V tile: vector load, transposed scalar stores -> Vt[d][key]
    for (int i = tid; i < 512; i += 256) {
      int r  = i >> 4;          // key
      int c8 = (i & 15) << 3;   // d base
      union { uint4 q; __bf16 e[8]; } t;
      t.q = *(const uint4*)&V[(headBase + t0 + r) * DHn + c8];
#pragma unroll
      for (int j = 0; j < 8; ++j) Vt[c8 + j][r] = t.e[j];
    }
#if USE_TDM
    if (wave == 0) __builtin_amdgcn_s_wait_tensorcnt(0);
#endif
    __syncthreads();

    // S = Q . K^T : two 16x16 frags over the 32 keys, K-contiguous B frags
    v8f s0 = vzero, s1 = vzero;
#pragma unroll
    for (int kc = 0; kc < 4; ++kc) {
      v16bf b0 = load_b16k(&Kt[0][kc * 32], 136);    // keys 0..15
      v16bf b1 = load_b16k(&Kt[16][kc * 32], 136);   // keys 16..31
      s0 = wmma_bf16(qf[kc], b0, s0);
      s1 = wmma_bf16(qf[kc], b1, s1);
    }

    // ---- online softmax (row stats replicated across each 16-lane half) ----
    float tm[8];
#pragma unroll
    for (int i = 0; i < 8; ++i) {
      s0[i] *= scale;
      s1[i] *= scale;
      tm[i] = fmaxf(s0[i], s1[i]);
    }
#pragma unroll
    for (int off = 1; off <= 8; off <<= 1)
#pragma unroll
      for (int i = 0; i < 8; ++i)
        tm[i] = fmaxf(tm[i], __shfl_xor(tm[i], off, 32));

    float fs[8];
#pragma unroll
    for (int i = 0; i < 8; ++i) {
      float mn = fmaxf(m8[i], tm[i]);
      fs[i] = __expf(m8[i] - mn);
      m8[i] = mn;
    }
    float p0[8], p1[8], rs[8];
#pragma unroll
    for (int i = 0; i < 8; ++i) {
      p0[i] = __expf(s0[i] - m8[i]);
      p1[i] = __expf(s1[i] - m8[i]);
      rs[i] = p0[i] + p1[i];
    }
#pragma unroll
    for (int off = 1; off <= 8; off <<= 1)
#pragma unroll
      for (int i = 0; i < 8; ++i) rs[i] += __shfl_xor(rs[i], off, 32);
#pragma unroll
    for (int i = 0; i < 8; ++i) l8[i] = l8[i] * fs[i] + rs[i];
#pragma unroll
    for (int df = 0; df < 8; ++df)
#pragma unroll
      for (int i = 0; i < 8; ++i) acc[df][i] *= fs[i];

    // C-layout -> A-layout via per-wave LDS staging of P (16x32 bf16)
#pragma unroll
    for (int i = 0; i < 8; ++i) {
      Pst[wave][rb + i][cc]      = (__bf16)p0[i];
      Pst[wave][rb + i][16 + cc] = (__bf16)p1[i];
    }
    v16bf pa = load_a16(&Pst[wave][0][0], 40);

    // O += P . V  (B frags K-contiguous from transposed Vt)
#pragma unroll
    for (int df = 0; df < 8; ++df) {
      v16bf bv = load_b16k(&Vt[df * 16][0], 40);
      acc[df] = wmma_bf16(pa, bv, acc[df]);
    }
  }

  // normalize and store O (row = b*S + s, col = h*DH + d)
  float inv[8];
#pragma unroll
  for (int i = 0; i < 8; ++i) inv[i] = 1.f / l8[i];
  const long orow0 = (long)b * Sn + qblk * 128 + wave * 16;
#pragma unroll
  for (int df = 0; df < 8; ++df)
#pragma unroll
    for (int i = 0; i < 8; ++i)
      O[(orow0 + rb + i) * Dn + h * DHn + df * 16 + cc] =
          (__bf16)(acc[df][i] * inv[i]);
}

// ---------------------------------------------------------------------------
// Host orchestration
// ---------------------------------------------------------------------------
extern "C" void kernel_launch(void* const* d_in, const int* in_sizes, int n_in,
                              void* d_out, int out_size, void* d_ws,
                              size_t ws_size, hipStream_t stream) {
  (void)in_sizes; (void)n_in; (void)out_size; (void)ws_size;
  const float* x    = (const float*)d_in[0];
  const float* cf   = (const float*)d_in[1];
  const float* W_Aq = (const float*)d_in[2];
  const float* W_Ak = (const float*)d_in[3];
  const float* W_Av = (const float*)d_in[4];
  const float* W_Bq = (const float*)d_in[5];
  const float* W_Bk = (const float*)d_in[6];
  const float* W_Bv = (const float*)d_in[7];
  const float* Wo   = (const float*)d_in[8];
  float* out = (float*)d_out;

  char* ws = (char*)d_ws;
  auto bump = [](size_t& o, size_t bytes) {
    size_t r = o;
    o += (bytes + 255) & ~(size_t)255;
    return r;
  };
  size_t off = 0;
  __bf16* xb  = (__bf16*)(ws + bump(off, (size_t)MR * Dn * 2));   // x bf16
  __bf16* Wc  = (__bf16*)(ws + bump(off, (size_t)Dn * NPP * 2));  // fused W (padded)
  __bf16* Wob = (__bf16*)(ws + bump(off, (size_t)Dn * Dn * 2));   // Wo bf16
  float*  Pf  = (float*) (ws + bump(off, (size_t)MR * NPP * 4));  // proj f32 (padded)
  __bf16* Qb  = (__bf16*)(ws + bump(off, (size_t)MR * Dn * 2));
  __bf16* Kb  = (__bf16*)(ws + bump(off, (size_t)MR * Dn * 2));
  __bf16* Vb  = (__bf16*)(ws + bump(off, (size_t)MR * Dn * 2));
  __bf16* Ob  = (__bf16*)(ws + bump(off, (size_t)MR * Dn * 2));

  // 1) conversions / packing
  {
    long n = MR * Dn;
    cvt_f32_bf16<<<(int)((n + 255) / 256), 256, 0, stream>>>(x, xb, n);
  }
  {
    long n = (long)Dn * Dn;
    cvt_f32_bf16<<<(int)((n + 255) / 256), 256, 0, stream>>>(Wo, Wob, n);
  }
  {
    long n = (long)Dn * NPP;  // zero whole fused-W (pad columns stay 0)
    fill_zero_bf16<<<(int)((n + 255) / 256), 256, 0, stream>>>(Wc, n);
  }
  auto packLaunch = [&](const float* w, int cols, int coff) {
    long n = (long)Dn * cols;
    pack_w<<<(int)((n + 255) / 256), 256, 0, stream>>>(w, Wc, cols, coff);
  };
  packLaunch(W_Aq, RQn * Hn, OFF_AQ);
  packLaunch(W_Ak, RKn * Hn, OFF_AK);
  packLaunch(W_Av, RVn * Hn, OFF_AV);
  packLaunch(W_Bq, RQn * DHn, OFF_BQ);
  packLaunch(W_Bk, RKn * DHn, OFF_BK);
  packLaunch(W_Bv, RVn * DHn, OFF_BV);

  // 2) fused projection GEMM: Pf[8192,1536] = xb @ Wc  (pad cols compute to 0)
  {
    dim3 grid(NPP / 128, (int)(MR / 128));
    gemm_bf16_f32<<<grid, 256, 0, stream>>>(xb, Wc, Pf, (int)MR, NPP, Dn);
  }

  // 3) RoPE + rank contraction -> Q/K/V (B,H,S,DH) bf16
  rope_qkv<<<(int)MR, 256, 0, stream>>>(Pf, cf, Qb, Kb, Vb);

  // 4) flash attention -> Ob (B*S, D) bf16
  flash_attn<<<Bn * Hn * (Sn / 128), 256, 0, stream>>>(Qb, Kb, Vb, Ob);

  // 5) output projection: out[8192,2048] = Ob @ Wob
  {
    dim3 grid(Dn / 128, (int)(MR / 128));
    gemm_bf16_f32<<<grid, 256, 0, stream>>>(Ob, Wob, out, (int)MR, Dn, Dn);
  }
}
